// HyperbolicTransformer_71545565216807
// MI455X (gfx1250) — matrix operand order repaired and checked by reading, not compiled, not run on hardware
//
#include <hip/hip_runtime.h>

// ---------------------------------------------------------------------------
// HyperbolicTransformer forward for MI455X (gfx1250, wave32, WMMA bf16,
// async global->LDS tile staging)
// ---------------------------------------------------------------------------

#define B_   16
#define L_   512
#define D_   512
#define NL_  4
#define V_   32000
#define C_   1000
#define H_   8
#define HD_  64
#define M_   (B_ * L_)       // 8192 rows
#define F_   (4 * D_)        // 2048 ffn dim

#define EPSV      1e-7f
#define MAX_NORM  (1.0f - 1e-5f)
#define LN_EPS    1e-5f

#ifndef __has_builtin
#define __has_builtin(x) 0
#endif

#define GLB_AS __attribute__((address_space(1)))
#define LDS_AS __attribute__((address_space(3)))

#if __has_builtin(__builtin_amdgcn_global_load_async_to_lds_b128)
#define HAVE_ASYNC_LDS 1
#else
#define HAVE_ASYNC_LDS 0
#endif

// 16-byte int vector types matching the async-LDS builtin's signature
typedef int v4i __attribute__((vector_size(16)));
typedef GLB_AS v4i gv4i;     // global-addrspace int4
typedef LDS_AS v4i lv4i;     // LDS-addrspace int4

typedef __attribute__((ext_vector_type(16))) __bf16 bf16x16;
typedef __attribute__((ext_vector_type(8)))  float  v8f;

union BFrag { bf16x16 v; unsigned int u[8]; };

static __device__ __forceinline__ unsigned short f2bf(float f) {
  union { float f; unsigned u; } c; c.f = f;
  unsigned r = (c.u + 0x7FFFu + ((c.u >> 16) & 1u)) >> 16;   // RNE
  return (unsigned short)r;
}
static __device__ __forceinline__ unsigned int pack2bf(float a, float b) {
  return (unsigned int)f2bf(a) | ((unsigned int)f2bf(b) << 16);
}
static __device__ __forceinline__ float geluf(float x) {
  return 0.5f * x * (1.0f + erff(x * 0.70710678118654752f));
}

static __device__ __forceinline__ void asyncWait0() {
#if HAVE_ASYNC_LDS
#if __has_builtin(__builtin_amdgcn_s_wait_asynccnt)
  __builtin_amdgcn_s_wait_asynccnt(0);
#else
  asm volatile("s_wait_asynccnt 0x0" ::: "memory");
#endif
#endif
}

// -------------------------- block reduction helper -------------------------
template<int NT>
static __device__ __forceinline__ float blockSum(float v, float* sbuf, int tid) {
  sbuf[tid] = v; __syncthreads();
  #pragma unroll
  for (int s = NT / 2; s > 0; s >>= 1) {
    if (tid < s) sbuf[tid] += sbuf[tid + s];
    __syncthreads();
  }
  float r = sbuf[0]; __syncthreads();
  return r;
}

// -------------------------- f32 -> bf16 convert ----------------------------
__global__ void k_convert_bf16(const float* __restrict__ src,
                               unsigned short* __restrict__ dst, int n) {
  for (int i = blockIdx.x * blockDim.x + threadIdx.x; i < n;
       i += gridDim.x * blockDim.x)
    dst[i] = f2bf(src[i]);
}

// -------------------------- embed + Poincare project -----------------------
__global__ void k_embed_project(const int* __restrict__ x,
                                const float* __restrict__ emb,
                                const float* __restrict__ pos,
                                float* __restrict__ h) {
  __shared__ float sbuf[128];
  int row = blockIdx.x;                 // 0..M-1
  int l   = row & (L_ - 1);
  int tok = x[row];
  int tid = threadIdx.x;
  float v[4]; float ss = 0.f;
  #pragma unroll
  for (int i = 0; i < 4; ++i) {
    int d = tid + i * 128;
    v[i] = emb[(size_t)tok * D_ + d] + pos[l * D_ + d];
    ss += v[i] * v[i];
  }
  float nrm2 = blockSum<128>(ss, sbuf, tid);
  float nrm  = sqrtf(nrm2);
  float sc   = (nrm < MAX_NORM) ? 1.0f : MAX_NORM / fmaxf(nrm, EPSV);
  #pragma unroll
  for (int i = 0; i < 4; ++i) h[(size_t)row * D_ + tid + i * 128] = v[i] * sc;
}

// -------------------------- residual add + project (generic rows) ----------
__global__ void k_project_rows(float* __restrict__ dst,
                               const float* __restrict__ src,
                               const float* __restrict__ delta) {
  __shared__ float sbuf[128];
  int row = blockIdx.x;
  int tid = threadIdx.x;
  float v[4]; float ss = 0.f;
  #pragma unroll
  for (int i = 0; i < 4; ++i) {
    int d = tid + i * 128;
    float t = src[(size_t)row * D_ + d];
    if (delta) t += delta[(size_t)row * D_ + d];
    v[i] = t; ss += t * t;
  }
  float nrm2 = blockSum<128>(ss, sbuf, tid);
  float nrm  = sqrtf(nrm2);
  float sc   = (nrm < MAX_NORM) ? 1.0f : MAX_NORM / fmaxf(nrm, EPSV);
  #pragma unroll
  for (int i = 0; i < 4; ++i) dst[(size_t)row * D_ + tid + i * 128] = v[i] * sc;
}

// -------------------------- layernorm -> bf16 ------------------------------
__global__ void k_layernorm_bf16(const float* __restrict__ x,
                                 const float* __restrict__ g,
                                 const float* __restrict__ b,
                                 unsigned short* __restrict__ out) {
  __shared__ float sbuf[256];
  int row = blockIdx.x, tid = threadIdx.x;
  float v0 = x[(size_t)row * D_ + tid];
  float v1 = x[(size_t)row * D_ + tid + 256];
  float s  = blockSum<256>(v0 + v1, sbuf, tid);
  float m  = s * (1.0f / D_);
  float s2 = blockSum<256>(v0 * v0 + v1 * v1, sbuf, tid);
  float var = s2 * (1.0f / D_) - m * m;
  float rstd = rsqrtf(var + LN_EPS);
  out[(size_t)row * D_ + tid]       = f2bf((v0 - m) * rstd * g[tid] + b[tid]);
  out[(size_t)row * D_ + tid + 256] = f2bf((v1 - m) * rstd * g[tid + 256] + b[tid + 256]);
}

// -------------------------- tiled WMMA bf16 GEMM ---------------------------
// C[M,N] = A[M,K](bf16) * Bw[K,N](bf16) + bias ; ACT: 0 none, 1 gelu
// BM=128, BN=64, BK=32, 8 waves/block, each wave: 16 rows x 64 cols (4 wmma)
// Tiles staged via GLOBAL_LOAD_ASYNC_TO_LDS_B128 (ASYNCcnt) when available.
template<int ACT, bool OUTBF>
__global__ __launch_bounds__(256)
void k_gemm_bf16(const unsigned short* __restrict__ A,
                 const unsigned short* __restrict__ Bw,
                 const float* __restrict__ bias,
                 float* __restrict__ outF,
                 unsigned short* __restrict__ outBF,
                 int M, int N, int K) {
  __shared__ __align__(16) unsigned int lAu[128 * 16];   // A tile 128x32 bf16
  __shared__ __align__(16) unsigned int lBu[32 * 32];    // B tile 32x64 bf16 (straight)
  unsigned short* lBs = (unsigned short*)lBu;

  int tid  = threadIdx.x;
  int lane = tid & 31, w = tid >> 5;
  int mBase = blockIdx.y * 128;
  int nBase = blockIdx.x * 64;
  int m    = lane & 15, half = lane >> 4;

  const unsigned int* Au = (const unsigned int*)A;

  v8f acc[4];
  #pragma unroll
  for (int s = 0; s < 4; ++s) acc[s] = (v8f)0.0f;

  for (int k0 = 0; k0 < K; k0 += 32) {
#if HAVE_ASYNC_LDS
    {
      // A tile: 128 rows x 64B = 512 b128 transfers (2 per lane)
      GLB_AS char* Ag = (GLB_AS char*)A;
      GLB_AS char* Bg = (GLB_AS char*)Bw;
      LDS_AS char* Al = (LDS_AS char*)lAu;
      LDS_AS char* Bl = (LDS_AS char*)lBu;
      #pragma unroll
      for (int i = 0; i < 2; ++i) {
        int idx = tid + i * 256;
        int r = idx >> 2, seg = idx & 3;
        __builtin_amdgcn_global_load_async_to_lds_b128(
            (gv4i*)(Ag + ((size_t)(mBase + r) * K + k0 + seg * 8) * 2),
            (lv4i*)(Al + (r * 32 + seg * 8) * 2), 0, 0);
      }
      // B tile: 32 rows x 128B = 256 b128 transfers (1 per lane)
      {
        int r = tid >> 3, seg = tid & 7;
        __builtin_amdgcn_global_load_async_to_lds_b128(
            (gv4i*)(Bg + ((size_t)(k0 + r) * N + nBase + seg * 8) * 2),
            (lv4i*)(Bl + (r * 64 + seg * 8) * 2), 0, 0);
      }
      asyncWait0();
    }
#else
    // fallback: manual staging through VGPRs
    #pragma unroll
    for (int i = 0; i < 8; ++i) {
      int idx = tid + i * 256;
      int r = idx >> 4, c2 = idx & 15;
      lAu[idx] = Au[((size_t)(mBase + r) * K + k0) / 2 + c2];
    }
    #pragma unroll
    for (int i = 0; i < 8; ++i) {
      int idx = tid + i * 256;
      int k = idx >> 6, n = idx & 63;
      lBs[k * 64 + n] = Bw[(size_t)(k0 + k) * N + nBase + n];
    }
#endif
    if (k0 + 32 < K)   // warm L2/GL1 for the next A tile
      __builtin_prefetch(&Au[((size_t)(mBase + (tid >> 4)) * K + k0 + 32) / 2], 0, 1);
    __syncthreads();

    // A fragment for this wave's 16 rows
    BFrag af;
    #pragma unroll
    for (int v = 0; v < 8; ++v) {
      int kp = ((v < 4) ? 0 : 16) + half * 8 + (v & 3) * 2;
      af.u[v] = lAu[(w * 16 + m) * 16 + (kp >> 1)];
    }
    // 4 column sub-tiles
    #pragma unroll
    for (int s = 0; s < 4; ++s) {
      BFrag bf;
      #pragma unroll
      for (int v = 0; v < 8; ++v) {
        int k = ((v < 4) ? 0 : 16) + half * 8 + (v & 3) * 2;
        unsigned int lo = lBs[k * 64 + s * 16 + m];
        unsigned int hi = lBs[(k + 1) * 64 + s * 16 + m];
        bf.u[v] = lo | (hi << 16);
      }
      acc[s] = __builtin_amdgcn_wmma_f32_16x16x32_bf16(
          false, af.v, false, bf.v, (short)0, acc[s], false, false);
    }
    __syncthreads();
  }

  // epilogue
  #pragma unroll
  for (int s = 0; s < 4; ++s) {
    #pragma unroll
    for (int v = 0; v < 8; ++v) {
      int row = mBase + w * 16 + v + 8 * half;
      int col = nBase + s * 16 + m;
      float val = acc[s][v] + bias[col];
      if (ACT == 1) val = geluf(val);
      if (OUTBF) outBF[(size_t)row * N + col] = f2bf(val);
      else       outF [(size_t)row * N + col] = val;
    }
  }
}

// ------------- per-head Poincare projection -> bf16 (+ optional ||k||^2) ---
__global__ void k_project_head_bf16(const float* __restrict__ src,
                                    unsigned short* __restrict__ dst,
                                    float* __restrict__ k2 /*nullable*/) {
  int lane = threadIdx.x & 31;
  int wv   = blockIdx.x * (blockDim.x >> 5) + (threadIdx.x >> 5); // vec id
  int row  = wv >> 3, h = wv & 7;
  size_t base = (size_t)row * D_ + h * HD_ + lane * 2;
  float a = src[base], b = src[base + 1];
  float ss = a * a + b * b;
  #pragma unroll
  for (int m = 16; m > 0; m >>= 1) ss += __shfl_xor(ss, m, 32);
  float nrm = sqrtf(ss);
  float sc  = (nrm < MAX_NORM) ? 1.0f : MAX_NORM / fmaxf(nrm, EPSV);
  ((unsigned int*)dst)[base >> 1] = pack2bf(a * sc, b * sc);
  if (k2 && lane == 0) k2[row * H_ + h] = ss * sc * sc;
}

// -------------------------- flash attention (one wave / 16 queries) --------
__global__ __launch_bounds__(32)
void k_attention(const unsigned short* __restrict__ qb,
                 const unsigned short* __restrict__ kb,
                 const unsigned short* __restrict__ vb,
                 const float* __restrict__ k2,
                 unsigned short* __restrict__ out) {
  __shared__ __align__(16) unsigned int lP[16 * 16];    // 16x32 bf16 P tile
  unsigned short* lPs = (unsigned short*)lP;

  int qt = blockIdx.x, h = blockIdx.y, b = blockIdx.z;
  int lane = threadIdx.x, m = lane & 15, half = lane >> 4;
  int bL = b * L_;
  int qrow0 = bL + qt * 16;
  const float invtemp = 0.125f;                   // 1/sqrt(64)

  const unsigned int* qbu = (const unsigned int*)qb;
  const unsigned int* kbu = (const unsigned int*)kb;

  // Q fragments (16x64 bf16 -> 2 A fragments of K=32)
  BFrag aq[2];
  #pragma unroll
  for (int g = 0; g < 2; ++g)
    #pragma unroll
    for (int v = 0; v < 8; ++v) {
      int kk = g * 32 + ((v < 4) ? 0 : 16) + half * 8 + (v & 3) * 2;
      aq[g].u[v] = qbu[((size_t)(qrow0 + m) * D_ + h * HD_ + kk) >> 1];
    }

  v8f acc_o[4];
  #pragma unroll
  for (int s = 0; s < 4; ++s) acc_o[s] = (v8f)0.0f;
  float mrow[8], lrow[8];
  #pragma unroll
  for (int v = 0; v < 8; ++v) { mrow[v] = -INFINITY; lrow[v] = 0.f; }

  for (int kc = 0; kc < L_ / 32; ++kc) {
    int key0 = kc * 32;
    float sc0[8], sc1[8];
    #pragma unroll
    for (int sub = 0; sub < 2; ++sub) {
      int key = key0 + sub * 16 + m;
      v8f sacc = (v8f)0.0f;
      #pragma unroll
      for (int g = 0; g < 2; ++g) {
        BFrag bk;
        #pragma unroll
        for (int v = 0; v < 8; ++v) {
          int d = g * 32 + ((v < 4) ? 0 : 16) + half * 8 + (v & 3) * 2;
          bk.u[v] = kbu[((size_t)(bL + key) * D_ + h * HD_ + d) >> 1];
        }
        sacc = __builtin_amdgcn_wmma_f32_16x16x32_bf16(
            false, aq[g].v, false, bk.v, (short)0, sacc, false, false);
      }
      float k2l = k2[(bL + key) * H_ + h];
      #pragma unroll
      for (int v = 0; v < 8; ++v) {
        float s = (2.0f * sacc[v] - k2l) * invtemp;
        if (sub == 0) sc0[v] = s; else sc1[v] = s;
      }
    }
    // online softmax update (per-row stats live in lane halves)
    #pragma unroll
    for (int v = 0; v < 8; ++v) {
      float mx = fmaxf(sc0[v], sc1[v]);
      #pragma unroll
      for (int msk = 8; msk > 0; msk >>= 1) mx = fmaxf(mx, __shfl_xor(mx, msk, 32));
      float nm = fmaxf(mrow[v], mx);
      float scale = __expf(mrow[v] - nm);
      float p0 = __expf(sc0[v] - nm), p1 = __expf(sc1[v] - nm);
      float rs = p0 + p1;
      #pragma unroll
      for (int msk = 8; msk > 0; msk >>= 1) rs += __shfl_xor(rs, msk, 32);
      lrow[v] = lrow[v] * scale + rs;
      mrow[v] = nm;
      #pragma unroll
      for (int s = 0; s < 4; ++s) acc_o[s][v] *= scale;
      // stash P in LDS (D-layout -> row-major 16x32 bf16)
      int row = v + 8 * half;
      lPs[row * 32 + m]      = f2bf(p0);
      lPs[row * 32 + 16 + m] = f2bf(p1);
    }
    __syncthreads();
    // reload P as A fragment
    BFrag pf;
    #pragma unroll
    for (int v = 0; v < 8; ++v) {
      int kp = ((v < 4) ? 0 : 16) + half * 8 + (v & 3) * 2;
      pf.u[v] = lP[m * 16 + (kp >> 1)];
    }
    // P(16x32) @ V(32x64) -> 4 wmma
    #pragma unroll
    for (int s = 0; s < 4; ++s) {
      BFrag bv;
      #pragma unroll
      for (int v = 0; v < 8; ++v) {
        int k = ((v < 4) ? 0 : 16) + half * 8 + (v & 3) * 2;
        int dim = h * HD_ + s * 16 + m;
        unsigned int lo = vb[(size_t)(bL + key0 + k)     * D_ + dim];
        unsigned int hi = vb[(size_t)(bL + key0 + k + 1) * D_ + dim];
        bv.u[v] = lo | (hi << 16);
      }
      acc_o[s] = __builtin_amdgcn_wmma_f32_16x16x32_bf16(
          false, pf.v, false, bv.v, (short)0, acc_o[s], false, false);
    }
    __syncthreads();
  }
  // write attention output (bf16, feeds Wo GEMM)
  #pragma unroll
  for (int s = 0; s < 4; ++s)
    #pragma unroll
    for (int v = 0; v < 8; ++v) {
      int row = qrow0 + v + 8 * half;
      int dim = h * HD_ + s * 16 + m;
      out[(size_t)row * D_ + dim] = f2bf(acc_o[s][v] / lrow[v]);
    }
}

// -------------------------- mean pool over L -------------------------------
__global__ void k_meanpool(const float* __restrict__ h, float* __restrict__ pooled) {
  int g = blockIdx.x * blockDim.x + threadIdx.x;      // B*D
  int b = g / D_, d = g % D_;
  float s = 0.f;
  for (int l = 0; l < L_; ++l) s += h[(size_t)(b * L_ + l) * D_ + d];
  pooled[g] = s * (1.0f / L_);
}

// -------------------------- hyperbolic distance ----------------------------
__global__ void k_distance(const float* __restrict__ p,   // [B,D] projected
                           const float* __restrict__ q,   // [C,D] projected
                           float* __restrict__ out) {     // [B,C] = -d
  int lane = threadIdx.x & 31;
  int idx  = blockIdx.x * (blockDim.x >> 5) + (threadIdx.x >> 5);
  int b = idx / C_, c = idx % C_;
  float diff2 = 0.f, p2 = 0.f, q2 = 0.f;
  for (int d = lane; d < D_; d += 32) {
    float pv = p[(size_t)b * D_ + d], qv = q[(size_t)c * D_ + d];
    float df = pv - qv;
    diff2 += df * df; p2 += pv * pv; q2 += qv * qv;
  }
  #pragma unroll
  for (int m = 16; m > 0; m >>= 1) {
    diff2 += __shfl_xor(diff2, m, 32);
    p2    += __shfl_xor(p2, m, 32);
    q2    += __shfl_xor(q2, m, 32);
  }
  if (lane == 0) {
    float denom = fmaxf((1.0f - p2) * (1.0f - q2), EPSV);
    float arg   = fmaxf(1.0f + 2.0f * diff2 / denom, 1.0f + EPSV);
    out[b * C_ + c] = -acoshf(arg);
  }
}

// ===========================================================================
extern "C" void kernel_launch(void* const* d_in, const int* in_sizes, int n_in,
                              void* d_out, int out_size, void* d_ws, size_t ws_size,
                              hipStream_t stream) {
  (void)in_sizes; (void)n_in; (void)out_size; (void)ws_size;
  const int*   x      = (const int*)  d_in[0];
  const float* emb    = (const float*)d_in[1];
  const float* pos    = (const float*)d_in[2];
  const float* protos = (const float*)d_in[3];
  const float* Wq = (const float*)d_in[4],  *bq = (const float*)d_in[5];
  const float* Wk = (const float*)d_in[6],  *bk = (const float*)d_in[7];
  const float* Wv = (const float*)d_in[8],  *bv = (const float*)d_in[9];
  const float* Wo = (const float*)d_in[10], *bo = (const float*)d_in[11];
  const float* g1 = (const float*)d_in[12], *be1 = (const float*)d_in[13];
  const float* g2 = (const float*)d_in[14], *be2 = (const float*)d_in[15];
  const float* W1 = (const float*)d_in[16], *b1 = (const float*)d_in[17];
  const float* W2 = (const float*)d_in[18], *b2 = (const float*)d_in[19];
  float* out = (float*)d_out;

  // ---------------- workspace carving (bytes) ----------------
  char* ws = (char*)d_ws;
  const size_t SZ_WDD = (size_t)NL_ * D_ * D_;      // per tensor elems
  const size_t SZ_WDF = (size_t)NL_ * D_ * F_;
  unsigned short* wqB = (unsigned short*)(ws);                       // 2MB each
  unsigned short* wkB = wqB + SZ_WDD;
  unsigned short* wvB = wkB + SZ_WDD;
  unsigned short* woB = wvB + SZ_WDD;
  unsigned short* w1B = woB + SZ_WDD;                                // 8MB
  unsigned short* w2B = w1B + SZ_WDF;                                // 8MB
  char* p = (char*)(w2B + SZ_WDF);                                   // 24MB in
  float* hbuf = (float*)p;             p += (size_t)M_ * D_ * 4;     // 16MB
  unsigned short* xnbf = (unsigned short*)p; p += (size_t)M_ * D_ * 2; // 8MB
  float* qf = (float*)p;               p += (size_t)M_ * D_ * 4;
  float* kf = (float*)p;               p += (size_t)M_ * D_ * 4;
  float* vf = (float*)p;               p += (size_t)M_ * D_ * 4;     // 48MB
  unsigned short* qb2 = (unsigned short*)p;                          // 32MB region
  unsigned short* kb2 = qb2 + (size_t)M_ * D_;
  unsigned short* vb2 = kb2 + (size_t)M_ * D_;
  unsigned short* geluB = qb2;                                       // reuse (M*F bf16)
  p += (size_t)M_ * F_ * 2;
  float* delta = (float*)p;            p += (size_t)M_ * D_ * 4;     // 16MB
  float* k2buf = (float*)p;            p += (size_t)M_ * H_ * 4;
  float* pooled = (float*)p;           p += (size_t)B_ * D_ * 4;
  float* protP  = (float*)p;           p += (size_t)C_ * D_ * 4;

  // ---------------- one-time weight conversion to bf16 ----------------
  k_convert_bf16<<<2048, 256, 0, stream>>>(Wq, wqB, (int)SZ_WDD);
  k_convert_bf16<<<2048, 256, 0, stream>>>(Wk, wkB, (int)SZ_WDD);
  k_convert_bf16<<<2048, 256, 0, stream>>>(Wv, wvB, (int)SZ_WDD);
  k_convert_bf16<<<2048, 256, 0, stream>>>(Wo, woB, (int)SZ_WDD);
  k_convert_bf16<<<4096, 256, 0, stream>>>(W1, w1B, (int)SZ_WDF);
  k_convert_bf16<<<4096, 256, 0, stream>>>(W2, w2B, (int)SZ_WDF);

  // ---------------- embedding + projection ----------------
  k_embed_project<<<M_, 128, 0, stream>>>(x, emb, pos, hbuf);

  dim3 gmmDD(D_ / 64, M_ / 128);       // N=512 GEMMs
  dim3 gmmDF(F_ / 64, M_ / 128);       // N=2048 GEMM
  dim3 gattn(L_ / 16, H_, B_);

  for (int i = 0; i < NL_; ++i) {
    const unsigned short* wq_i = wqB + (size_t)i * D_ * D_;
    const unsigned short* wk_i = wkB + (size_t)i * D_ * D_;
    const unsigned short* wv_i = wvB + (size_t)i * D_ * D_;
    const unsigned short* wo_i = woB + (size_t)i * D_ * D_;
    const unsigned short* w1_i = w1B + (size_t)i * D_ * F_;
    const unsigned short* w2_i = w2B + (size_t)i * F_ * D_;

    k_layernorm_bf16<<<M_, 256, 0, stream>>>(hbuf, g1 + i * D_, be1 + i * D_, xnbf);

    k_gemm_bf16<0,false><<<gmmDD, 256, 0, stream>>>(xnbf, wq_i, bq + i * D_, qf, nullptr, M_, D_, D_);
    k_gemm_bf16<0,false><<<gmmDD, 256, 0, stream>>>(xnbf, wk_i, bk + i * D_, kf, nullptr, M_, D_, D_);
    k_gemm_bf16<0,false><<<gmmDD, 256, 0, stream>>>(xnbf, wv_i, bv + i * D_, vf, nullptr, M_, D_, D_);

    k_project_head_bf16<<<M_ * H_ / 8, 256, 0, stream>>>(qf, qb2, nullptr);
    k_project_head_bf16<<<M_ * H_ / 8, 256, 0, stream>>>(kf, kb2, k2buf);
    k_convert_bf16<<<4096, 256, 0, stream>>>(vf, vb2, M_ * D_);

    k_attention<<<gattn, 32, 0, stream>>>(qb2, kb2, vb2, k2buf, xnbf);

    k_gemm_bf16<0,false><<<gmmDD, 256, 0, stream>>>(xnbf, wo_i, bo + i * D_, delta, nullptr, M_, D_, D_);
    k_project_rows<<<M_, 128, 0, stream>>>(hbuf, hbuf, delta);

    k_layernorm_bf16<<<M_, 256, 0, stream>>>(hbuf, g2 + i * D_, be2 + i * D_, xnbf);
    k_gemm_bf16<1,true ><<<gmmDF, 256, 0, stream>>>(xnbf, w1_i, b1 + i * F_, nullptr, geluB, M_, F_, D_);
    k_gemm_bf16<0,false><<<gmmDD, 256, 0, stream>>>(geluB, w2_i, b2 + i * D_, delta, nullptr, M_, D_, F_);
    k_project_rows<<<M_, 128, 0, stream>>>(hbuf, hbuf, delta);
  }

  // ---------------- pooling, projection, hyperbolic distance ---------------
  k_meanpool<<<B_ * D_ / 256, 256, 0, stream>>>(hbuf, pooled);
  k_project_rows<<<B_, 128, 0, stream>>>(pooled, pooled, nullptr);
  k_project_rows<<<C_, 128, 0, stream>>>(protP, protos, nullptr);
  k_distance<<<B_ * C_ / 8, 256, 0, stream>>>(pooled, protP, out);
}
